// SelfAttnFunc_81647328297665
// MI455X (gfx1250) — compile-verified, matrix operand order checked by reading, not run on hardware
//
#include <hip/hip_runtime.h>
#include <hip/hip_bf16.h>
#include <math.h>
#include <stdint.h>

typedef __attribute__((ext_vector_type(16))) __bf16 v16bf;
typedef __attribute__((ext_vector_type(8)))  __bf16 v8bf;
typedef __attribute__((ext_vector_type(8)))  float  v8f;
typedef __attribute__((ext_vector_type(4)))  unsigned int u32x4;
typedef __attribute__((ext_vector_type(4)))  int i32x4;
typedef __attribute__((ext_vector_type(8)))  int i32x8;

#define WMMA_BF16(a, b, c) \
  __builtin_amdgcn_wmma_f32_16x16x32_bf16(false, (a), false, (b), (short)0, (c), false, false)

static constexpr int BQ    = 8;
static constexpr int SEQ   = 1024;
static constexpr int DIM   = 1024;
static constexpr int NH    = 16;
static constexpr int HD    = 64;          // head dim
static constexpr int M     = BQ * SEQ;    // 8192 token rows
static constexpr int N_QKV = 3 * DIM;     // 3072

// ---------------- f32 -> bf16 convert ----------------
__global__ void cvt_f32_bf16(const float* __restrict__ in, __bf16* __restrict__ out, int n) {
  int i = blockIdx.x * blockDim.x + threadIdx.x;
  if (i < n) out[i] = (__bf16)in[i];
}

// Load one 16x32 bf16 operand fragment in the CDNA5 WMMA A/B lane layout.
// lane L: row = row0 + (L&15); holds K = {8*(L>>4)+0..7} and {16+8*(L>>4)+0..7}.
__device__ inline v16bf load_frag(const __bf16* __restrict__ base, int ld, int row0,
                                  int k0, int lane) {
  int r = row0 + (lane & 15);
  int h = lane >> 4;
  const __bf16* p = base + (size_t)r * ld + k0 + 8 * h;
  v8bf lo = *(const v8bf*)(p);
  v8bf hi = *(const v8bf*)(p + 16);
  return __builtin_shufflevector(lo, hi, 0, 1, 2, 3, 4, 5, 6, 7,
                                 8, 9, 10, 11, 12, 13, 14, 15);
}

// ---------------- Tensor Data Mover: 2D bf16 tile -> LDS ----------------
// D# per CDNA5 ISA. Tile is fully in-bounds, so
// tensor dims == tile dims. Strides/dims in data_size (=2B) units.
__device__ inline void tdm_load_2d_bf16(unsigned lds_byte_off, const void* gptr,
                                        unsigned tile_w, unsigned tile_h,
                                        unsigned row_stride_elems) {
  unsigned long long ga = (unsigned long long)(uintptr_t)gptr;
  u32x4 g0;
  g0[0] = 1u;                                         // count=1, is_restore=0
  g0[1] = lds_byte_off;                               // lds_addr [63:32]
  g0[2] = (unsigned)(ga & 0xffffffffu);               // global_addr[31:0]
  g0[3] = (unsigned)((ga >> 32) & 0x1ffffffu)         // global_addr[56:32]
        | (2u << 30);                                 // type=2 ("image")

  union { unsigned long long w[4]; i32x8 v; } g1;
  g1.w[0] = (1ull << 16)                              // data_size=1 (2 bytes)
          | ((unsigned long long)(tile_w & 0xffffu) << 48);   // tensor_dim0[15:0]
  g1.w[1] = ((unsigned long long)(tile_w >> 16) & 0xffffull)  // tensor_dim0[31:16]
          | ((unsigned long long)tile_h << 16)                // tensor_dim1 [111:80]
          | ((unsigned long long)(tile_w & 0xffffu) << 48);   // tile_dim0 [127:112]
  g1.w[2] = (unsigned long long)(tile_h & 0xffffu)            // tile_dim1 [143:128]
          | ((unsigned long long)row_stride_elems << 32);     // dim0_stride[31:0] @ [191:160]
  g1.w[3] = 0ull;                                             // dim0_stride[47:32]=0, dim1_stride=0

  i32x4 z4 = {0, 0, 0, 0};
  i32x8 z8 = {0, 0, 0, 0, 0, 0, 0, 0};
  // amdgpu-toolchain (clang-23) 6-arg form: (g0, g1, g2, g3, extra, cpol)
  __builtin_amdgcn_tensor_load_to_lds(g0, g1.v, z4, z4, z8, 0);
}

__device__ inline unsigned lds_off(const void* p) {
  return (unsigned)(uintptr_t)p;   // flat LDS addr: low 32 bits = LDS byte offset
}

// ---------------- GEMM: C = A * Bw^T + bias (TDM double-buffered) ----------
// A: [M,K] bf16 row-major.  Bw: [N,K] bf16 row-major (ref does X @ W.T).
// Block = 256 threads = 8 waves; block tile 128(M) x 64(N); wave tile 16 x 64.
// Wave 0 drives the TDM pipeline: tiles A(128x32) and B(64x32) per k-step.
template <bool OUT_BF16>
__global__ void __launch_bounds__(256) gemm_bf16_bias(
    const __bf16* __restrict__ A, const __bf16* __restrict__ Bw,
    const float* __restrict__ bias, void* __restrict__ Cout,
    int Mdim, int Ndim, int Kdim) {
  __shared__ __bf16 As[2][128 * 32];   // 2 x 8 KB
  __shared__ __bf16 Bs[2][64 * 32];    // 2 x 4 KB
  int wave = threadIdx.x >> 5;
  int lane = threadIdx.x & 31;
  int m0 = blockIdx.x * 128 + wave * 16;
  int n0 = blockIdx.y * 64;
  const __bf16* Ablk = A + (size_t)blockIdx.x * 128 * Kdim;
  const __bf16* Bblk = Bw + (size_t)n0 * Kdim;

  int nsteps = Kdim / 32;
  if (wave == 0) {
    tdm_load_2d_bf16(lds_off(&As[0][0]), Ablk, 32, 128, Kdim);
    tdm_load_2d_bf16(lds_off(&Bs[0][0]), Bblk, 32, 64, Kdim);
  }

  v8f acc[4] = {};
  for (int s = 0; s < nsteps; ++s) {
    int cur = s & 1;
    if (wave == 0) {
      if (s + 1 < nsteps) {
        tdm_load_2d_bf16(lds_off(&As[cur ^ 1][0]), Ablk + (s + 1) * 32, 32, 128, Kdim);
        tdm_load_2d_bf16(lds_off(&Bs[cur ^ 1][0]), Bblk + (s + 1) * 32, 32, 64, Kdim);
        __builtin_amdgcn_s_wait_tensorcnt(2);  // in-order: current pair complete
      } else {
        __builtin_amdgcn_s_wait_tensorcnt(0);
      }
    }
    __syncthreads();                           // publish TDM-written LDS to all waves
    v16bf a = load_frag(&As[cur][0], 32, wave * 16, 0, lane);
#pragma unroll
    for (int t = 0; t < 4; ++t) {
      v16bf b = load_frag(&Bs[cur][0], 32, t * 16, 0, lane);
      acc[t] = WMMA_BF16(a, b, acc[t]);
    }
    __syncthreads();                           // all reads done before buffer reuse
  }

  int half = lane >> 4, col = lane & 15;
#pragma unroll
  for (int t = 0; t < 4; ++t) {
    int n = n0 + 16 * t + col;
    float bv = bias[n];
#pragma unroll
    for (int v = 0; v < 8; ++v) {
      int row = m0 + v + 8 * half;          // C/D layout: M = vgpr + 8*(lane/16)
      float val = acc[t][v] + bv;
      if (OUT_BF16)
        ((__bf16*)Cout)[(size_t)row * Ndim + n] = (__bf16)val;
      else
        ((float*)Cout)[(size_t)row * Ndim + n] = val;
    }
  }
}

// ---------------- QKV rearrange ----------------
// qkv bf16 [B,S,3,H,hd] -> Q [B,H,S,hd], K [B,H,S,hd], Vt [B,H,hd,S]
__global__ void rearrange_qkv(const __bf16* __restrict__ qkv, __bf16* __restrict__ Q,
                              __bf16* __restrict__ K, __bf16* __restrict__ Vt) {
  int idx = blockIdx.x * blockDim.x + threadIdx.x;
  const int total = M * N_QKV;
  if (idx >= total) return;
  int d  = idx % HD;
  int h  = (idx / HD) % NH;
  int th = (idx / (HD * NH)) % 3;
  int s  = (idx / (HD * NH * 3)) % SEQ;
  int b  = idx / (HD * NH * 3 * SEQ);
  __bf16 v = qkv[idx];
  size_t bh = (size_t)(b * NH + h);
  if (th == 0)      Q[(bh * SEQ + s) * HD + d] = v;
  else if (th == 1) K[(bh * SEQ + s) * HD + d] = v;
  else              Vt[(bh * HD + d) * SEQ + s] = v;
}

// ---------------- Flash attention ----------------
// Block = 4 waves; wave handles a 16-query tile; key blocks of 32.
// grid.x = B*H, grid.y = S/64.
__global__ void __launch_bounds__(128) attn_kernel(
    const __bf16* __restrict__ Q, const __bf16* __restrict__ K,
    const __bf16* __restrict__ Vt, __bf16* __restrict__ Ctx) {
  __shared__ __bf16 Pl[4][16 * 32];   // per-wave 16x32 probability tile

  int wave = threadIdx.x >> 5;
  int lane = threadIdx.x & 31;
  int bh = blockIdx.x;                 // b*NH + h
  int q0 = blockIdx.y * 64 + wave * 16;
  int b = bh >> 4, h = bh & 15;
  int half = lane >> 4, col = lane & 15;

  const __bf16* Qh = Q + (size_t)bh * SEQ * HD;
  const __bf16* Kh = K + (size_t)bh * SEQ * HD;
  const __bf16* Vh = Vt + (size_t)bh * HD * SEQ;

  v16bf qf0 = load_frag(Qh, HD, q0, 0, lane);
  v16bf qf1 = load_frag(Qh, HD, q0, 32, lane);

  v8f acc[4] = {};
  float mrow[8], lrow[8];
#pragma unroll
  for (int v = 0; v < 8; ++v) { mrow[v] = -__builtin_inff(); lrow[v] = 0.f; }
  const float scale = 0.125f;  // 1/sqrt(64)

  for (int kb = 0; kb < SEQ; kb += 32) {
    // ---- scores: two 16(query) x 16(key) tiles over hd=64 (2 k-chunks) ----
    v8f s0 = {}, s1 = {};
    {
      v16bf k00 = load_frag(Kh, HD, kb, 0, lane);
      v16bf k10 = load_frag(Kh, HD, kb + 16, 0, lane);
      s0 = WMMA_BF16(qf0, k00, s0);
      s1 = WMMA_BF16(qf0, k10, s1);
      v16bf k01 = load_frag(Kh, HD, kb, 32, lane);
      v16bf k11 = load_frag(Kh, HD, kb + 16, 32, lane);
      s0 = WMMA_BF16(qf1, k01, s0);
      s1 = WMMA_BF16(qf1, k11, s1);
    }
    // ---- online softmax; row state replicated across each 16-lane half ----
    float p0[8], p1[8];
#pragma unroll
    for (int v = 0; v < 8; ++v) {
      float a0 = s0[v] * scale, a1 = s1[v] * scale;
      float rm = fmaxf(a0, a1);
      rm = fmaxf(rm, __shfl_xor(rm, 1, 32));
      rm = fmaxf(rm, __shfl_xor(rm, 2, 32));
      rm = fmaxf(rm, __shfl_xor(rm, 4, 32));
      rm = fmaxf(rm, __shfl_xor(rm, 8, 32));
      float mn = fmaxf(mrow[v], rm);
      float e0 = __expf(a0 - mn), e1 = __expf(a1 - mn);
      float rs = e0 + e1;
      rs += __shfl_xor(rs, 1, 32);
      rs += __shfl_xor(rs, 2, 32);
      rs += __shfl_xor(rs, 4, 32);
      rs += __shfl_xor(rs, 8, 32);
      float corr = __expf(mrow[v] - mn);
      lrow[v] = lrow[v] * corr + rs;
      mrow[v] = mn;
#pragma unroll
      for (int t = 0; t < 4; ++t) acc[t][v] *= corr;
      p0[v] = e0; p1[v] = e1;
    }
    // ---- D-layout P -> row-major LDS -> A-layout fragment ----
    __bf16* pl = Pl[wave];
#pragma unroll
    for (int v = 0; v < 8; ++v) {
      int m = v + 8 * half;
      pl[m * 32 + col]      = (__bf16)p0[v];
      pl[m * 32 + 16 + col] = (__bf16)p1[v];
    }
    __syncthreads();
    v16bf pf;
    {
      const __bf16* pp = pl + (size_t)col * 32 + 8 * half;
      v8bf lo = *(const v8bf*)(pp);
      v8bf hi = *(const v8bf*)(pp + 16);
      pf = __builtin_shufflevector(lo, hi, 0, 1, 2, 3, 4, 5, 6, 7,
                                   8, 9, 10, 11, 12, 13, 14, 15);
    }
    // ---- ctx += P(16x32 keys) x V(32 keys x 64 hd), Vt rows are hd ----
#pragma unroll
    for (int t = 0; t < 4; ++t) {
      v16bf vf = load_frag(Vh, SEQ, t * 16, kb, lane);
      acc[t] = WMMA_BF16(pf, vf, acc[t]);
    }
    __syncthreads();
  }

  // ---- normalize + write ctx as [B,S,D] bf16 (D = h*64 + d) ----
#pragma unroll
  for (int t = 0; t < 4; ++t) {
    int d = h * HD + t * 16 + col;
#pragma unroll
    for (int v = 0; v < 8; ++v) {
      int tok = q0 + v + 8 * half;
      float val = acc[t][v] / lrow[v];
      Ctx[((size_t)b * SEQ + tok) * DIM + d] = (__bf16)val;
    }
  }
}

extern "C" void kernel_launch(void* const* d_in, const int* in_sizes, int n_in,
                              void* d_out, int out_size, void* d_ws, size_t ws_size,
                              hipStream_t stream) {
  (void)in_sizes; (void)n_in; (void)out_size; (void)ws_size;
  const float* X  = (const float*)d_in[0];   // [8,1024,1024]
  const float* Wi = (const float*)d_in[1];   // [3072,1024]
  const float* Wo = (const float*)d_in[2];   // [1024,1024]
  const float* bi = (const float*)d_in[3];   // [3072]
  const float* bo = (const float*)d_in[4];   // [1024]
  float* out = (float*)d_out;                // [8,1024,1024] fp32

  char* ws = (char*)d_ws;
  size_t off = 0;
  auto alloc = [&](size_t bytes) -> void* {
    void* p = ws + off;
    off += (bytes + 255) & ~(size_t)255;
    return p;
  };
  __bf16* Xbf   = (__bf16*)alloc((size_t)M * DIM * 2);
  __bf16* Wibf  = (__bf16*)alloc((size_t)N_QKV * DIM * 2);
  __bf16* Wobf  = (__bf16*)alloc((size_t)DIM * DIM * 2);
  __bf16* QKVbf = (__bf16*)alloc((size_t)M * N_QKV * 2);
  __bf16* Qb    = (__bf16*)alloc((size_t)M * DIM * 2);
  __bf16* Kb    = (__bf16*)alloc((size_t)M * DIM * 2);
  __bf16* Vtb   = (__bf16*)alloc((size_t)M * DIM * 2);
  __bf16* Ctxbf = (__bf16*)alloc((size_t)M * DIM * 2);

  // 1) converts
  {
    int n = M * DIM;
    cvt_f32_bf16<<<(n + 255) / 256, 256, 0, stream>>>(X, Xbf, n);
    n = N_QKV * DIM;
    cvt_f32_bf16<<<(n + 255) / 256, 256, 0, stream>>>(Wi, Wibf, n);
    n = DIM * DIM;
    cvt_f32_bf16<<<(n + 255) / 256, 256, 0, stream>>>(Wo, Wobf, n);
  }
  // 2) fused QKV projection (TDM-staged, bias fused, bf16 out)
  gemm_bf16_bias<true><<<dim3(M / 128, N_QKV / 64), 256, 0, stream>>>(
      Xbf, Wibf, bi, QKVbf, M, N_QKV, DIM);
  // 3) split/rearrange (V transposed for the PV WMMA)
  {
    int n = M * N_QKV;
    rearrange_qkv<<<(n + 255) / 256, 256, 0, stream>>>(QKVbf, Qb, Kb, Vtb);
  }
  // 4) flash attention
  attn_kernel<<<dim3(BQ * NH, SEQ / 64), 128, 0, stream>>>(Qb, Kb, Vtb, Ctxbf);
  // 5) output projection (TDM-staged, fp32 out + bias)
  gemm_bf16_bias<false><<<dim3(M / 128, DIM / 64), 256, 0, stream>>>(
      Ctxbf, Wobf, bo, out, M, DIM, DIM);
}
